// MAlphaExactAttention_11398843203809
// MI455X (gfx1250) — matrix operand order repaired and checked by reading, not compiled, not run on hardware
//
#include <hip/hip_runtime.h>
#include <hip/hip_bf16.h>

// ---------------- problem constants ----------------
constexpr int NP    = 1024;          // patches (32x32 grid)
constexpr int DIMC  = 512;           // model dim
constexpr int NH    = 8;             // heads
constexpr int HD    = 64;            // head dim
constexpr int NB    = 8;             // batch
constexpr int TOKENS = NB * NP;      // 8192
constexpr float EPSV = 1e-6f;
constexpr int LDB   = 520;           // padded LDS row stride (halves): 1040B, 16B-aligned,
                                     // 1040 % 256 = 16 -> 16 fragment lanes hit 16 distinct banks

typedef __attribute__((ext_vector_type(16))) _Float16 v16h;
typedef __attribute__((ext_vector_type(8)))  float    v8f;

union FragU {
  v16h   h;
  float4 q[2];
};

#define WMMA_F16(A, B, C) \
  __builtin_amdgcn_wmma_f32_16x16x32_f16(false, (A), false, (B), (short)0, (C), false, false)

// ---- CDNA5 async global->LDS copy (ASYNCcnt-tracked, no VGPR staging) ----
__device__ __forceinline__ void async_copy_b128(unsigned lds_off, const void* gaddr) {
  asm volatile("global_load_async_to_lds_b128 %0, %1, off"
               :: "v"(lds_off), "v"(gaddr) : "memory");
}
__device__ __forceinline__ void wait_async0() {
  asm volatile("s_wait_asynccnt 0x0" ::: "memory");
}

// A-matrix 16x32 f16 fragment (ISA 7.12.2, 16-bit A table):
// lane l: m = l&15, half = l>>4; regs 0..3 hold K = {2r,2r+1}+8*half,
// regs 4..7 hold K = 16+{..}+8*half  => two 16B K-contiguous loads.
__device__ __forceinline__ v16h load_frag_a(const _Float16* __restrict__ base,
                                            int ld, int k0) {
  const int l  = threadIdx.x & 31;
  const int r  = l & 15;
  const int hf = l >> 4;
  const _Float16* p = base + r * ld + k0 + 8 * hf;
  FragU f;
  f.q[0] = *reinterpret_cast<const float4*>(p);
  f.q[1] = *reinterpret_cast<const float4*>(p + 16);
  return f.h;
}

// B-matrix 32x16 f16 fragment (matches A table order, K=32):
// lane l: n = l&15, g = l>>4; lane holds column n, K = 8g..8g+7 then 16+8g..16+8g+7.
__device__ __forceinline__ v16h load_frag_b(const _Float16* __restrict__ base,
                                            int ld, int k0) {
  const int l = threadIdx.x & 31;
  const int n = l & 15;
  const int g = l >> 4;
  const _Float16* p = base + n * ld + k0 + 8 * g;
  FragU f;
  f.q[0] = *reinterpret_cast<const float4*>(p);
  f.q[1] = *reinterpret_cast<const float4*>(p + 16);
  return f.h;
}

// ---------------- f32 -> f16 conversion ----------------
__global__ void cvt_f32_f16(const float* __restrict__ s, _Float16* __restrict__ d,
                            int nelem) {
  int i = blockIdx.x * blockDim.x + threadIdx.x;
  const int stride = gridDim.x * blockDim.x;
  for (; i < nelem; i += stride) d[i] = (_Float16)s[i];
}

// ---------------- mask = sum_k alphas[k] * W^k ----------------
__global__ void mask_kernel(const float* __restrict__ alphas,
                            const float* __restrict__ powers,
                            float* __restrict__ mask) {
  const int NN = NP * NP;
  int i = blockIdx.x * blockDim.x + threadIdx.x;
  const int stride = gridDim.x * blockDim.x;
  for (; i < NN; i += stride) {
    float s = 0.f;
#pragma unroll
    for (int k = 0; k < 6; ++k) s += alphas[k] * powers[k * NN + i];
    mask[i] = s;
  }
}

// ---- stage a 64-col x 512-K f16 weight tile into padded LDS via async copy ----
// wsrc points at a contiguous 64x512 slab (row o, K contiguous).
__device__ __forceinline__ void stage_weights(_Float16* shb,
                                              const _Float16* __restrict__ wsrc) {
  const unsigned base = (unsigned)(size_t)shb;   // low 32 bits = LDS offset
  const int tid = threadIdx.x;                   // 0..127
#pragma unroll
  for (int j = 0; j < 32; ++j) {
    const int e = j * 128 + tid;                 // 0..4095 16B-chunks
    const int r = e >> 6;                        // row  (0..63)
    const int c = e & 63;                        // chunk(0..63) of 8 halves
    async_copy_b128(base + (unsigned)(r * LDB + c * 8) * 2,
                    wsrc + (size_t)r * DIMC + c * 8);
  }
  wait_async0();
  __syncthreads();
}

// ---------------- QKV GEMM: (8192x512) @ (512x1536)^T ----------------
// 4 waves/block share an async-staged LDS weight tile (64 cols x 512 K).
// Each wave: 16 tokens x 64 cols, A double-buffered from global.
// Scatter-stores q/k (B,H,N,D) with ReLU+eps and v transposed (B,H,D,N).
__global__ void __launch_bounds__(128)
gemm_qkv(const _Float16* __restrict__ xh, const _Float16* __restrict__ wh,
         _Float16* __restrict__ qh, _Float16* __restrict__ kh,
         _Float16* __restrict__ vth) {
  __shared__ __align__(16) _Float16 shb[64 * LDB];
  const int o0 = blockIdx.y * 64;
  stage_weights(shb, wh + (size_t)o0 * DIMC);

  const int w  = threadIdx.x >> 5;
  const int t0 = blockIdx.x * 64 + w * 16;
  const _Float16* abase = xh + (size_t)t0 * DIMC;

  v8f acc[4] = {};
  v16h a = load_frag_a(abase, DIMC, 0);
#pragma unroll
  for (int kk = 0; kk < DIMC; kk += 32) {
    const v16h acur = a;
    if (kk + 32 < DIMC) a = load_frag_a(abase, DIMC, kk + 32);
#pragma unroll
    for (int t = 0; t < 4; ++t) {
      v16h bfr = load_frag_b(shb + 16 * t * LDB, LDB, kk);
      acc[t] = WMMA_F16(acur, bfr, acc[t]);
    }
  }

  const int l  = threadIdx.x & 31;
  const int n  = l & 15;
  const int mb = (l >> 4) * 8;      // C/D layout: lanes 16-31 own rows 8..15
#pragma unroll
  for (int t = 0; t < 4; ++t) {
    const int o     = o0 + 16 * t + n;
    const int which = o >> 9;       // 0:q 1:k 2:v
    const int c     = o & 511;
    const int hh    = c >> 6;
    const int d     = c & 63;
#pragma unroll
    for (int r = 0; r < 8; ++r) {
      const int tok = t0 + mb + r;
      const int bb  = tok >> 10;
      const int nn  = tok & 1023;
      const int bh  = bb * NH + hh;
      const float v = acc[t][r];
      if (which == 0)
        qh[(size_t)(bh * NP + nn) * HD + d] = (_Float16)(fmaxf(v, 0.f) + EPSV);
      else if (which == 1)
        kh[(size_t)(bh * NP + nn) * HD + d] = (_Float16)(fmaxf(v, 0.f) + EPSV);
      else
        vth[(size_t)(bh * HD + d) * NP + nn] = (_Float16)v;
    }
  }
}

// ---------------- streaming masked linear attention ----------------
// One wave per (b,h, 16-row Q tile). Streams keys in blocks of 32:
//   S = Q K^T (WMMA), masked = S * mask, rowsum += masked,
//   out += masked @ V (WMMA, A-frag rebuilt via LDS bounce).
// Finally out *= 1/(rowsum+eps), stored as f16 in (token, C) layout.
__global__ void __launch_bounds__(32)
attn_kernel(const _Float16* __restrict__ qh, const _Float16* __restrict__ kh,
            const _Float16* __restrict__ vth, const float* __restrict__ mask,
            _Float16* __restrict__ attnh) {
  __shared__ __align__(16) _Float16 sh[16][40];   // 16x32 masked tile (+pad)

  const int q0 = blockIdx.x * 16;
  const int bh = blockIdx.y;
  const int bb = bh >> 3;
  const int hh = bh & 7;

  const _Float16* qbase = qh + (size_t)(bh * NP + q0) * HD;
  const v16h qa0 = load_frag_a(qbase, HD, 0);
  const v16h qa1 = load_frag_a(qbase, HD, 32);

  v8f out[4] = {};
  v8f rowacc = {};

  const int l  = threadIdx.x & 31;
  const int n  = l & 15;
  const int mb = (l >> 4) * 8;
  const int hf = l >> 4;

  const _Float16* vbase = vth + (size_t)bh * HD * NP;

  for (int kb = 0; kb < NP; kb += 32) {
    // --- S tiles: 16 q-rows x 32 keys ---
    v8f s0 = {}, s1 = {};
    {
      const _Float16* kbase0 = kh + (size_t)(bh * NP + kb) * HD;
      v16h b0 = load_frag_b(kbase0, HD, 0);
      v16h b1 = load_frag_b(kbase0, HD, 32);
      s0 = WMMA_F16(qa0, b0, s0);
      s0 = WMMA_F16(qa1, b1, s0);
      const _Float16* kbase1 = kbase0 + 16 * HD;
      v16h b2 = load_frag_b(kbase1, HD, 0);
      v16h b3 = load_frag_b(kbase1, HD, 32);
      s1 = WMMA_F16(qa0, b2, s1);
      s1 = WMMA_F16(qa1, b3, s1);
    }
    // --- mask multiply, rowsum accumulate, redistribute via LDS ---
#pragma unroll
    for (int r = 0; r < 8; ++r) {
      const int row = q0 + mb + r;
      const float m0 = mask[(size_t)row * NP + kb + n];
      const float m1 = mask[(size_t)row * NP + kb + 16 + n];
      const float v0 = s0[r] * m0;
      const float v1 = s1[r] * m1;
      rowacc[r] += v0 + v1;
      sh[mb + r][n]      = (_Float16)v0;
      sh[mb + r][16 + n] = (_Float16)v1;
    }
    __syncthreads();
    FragU fa;
    {
      const int rr = l & 15;
      fa.q[0] = *reinterpret_cast<const float4*>(&sh[rr][8 * hf]);
      fa.q[1] = *reinterpret_cast<const float4*>(&sh[rr][16 + 8 * hf]);
    }
    // --- out += masked @ V ---
#pragma unroll
    for (int t = 0; t < 4; ++t) {
      v16h vb = load_frag_b(vbase + (size_t)(16 * t) * NP, NP, kb);
      out[t] = WMMA_F16(fa.h, vb, out[t]);
    }
    __syncthreads();
  }

  // --- normalize: reduce rowsum across the 16 lanes of each half ---
#pragma unroll
  for (int r = 0; r < 8; ++r) {
    float s = rowacc[r];
    s += __shfl_xor(s, 1, 32);
    s += __shfl_xor(s, 2, 32);
    s += __shfl_xor(s, 4, 32);
    s += __shfl_xor(s, 8, 32);
    rowacc[r] = 1.0f / (s + EPSV);
  }
  // --- scale and store to (token, C) f16 for the final projection ---
#pragma unroll
  for (int t = 0; t < 4; ++t) {
    const int d = 16 * t + n;
#pragma unroll
    for (int r = 0; r < 8; ++r) {
      const int tok = bb * NP + q0 + mb + r;
      attnh[(size_t)tok * DIMC + hh * HD + d] = (_Float16)(out[t][r] * rowacc[r]);
    }
  }
}

// ---------------- output projection: (8192x512) @ (512x512)^T + b ----------------
__global__ void __launch_bounds__(128)
gemm_out(const _Float16* __restrict__ ah, const _Float16* __restrict__ wh,
         const float* __restrict__ bias, float* __restrict__ outp) {
  __shared__ __align__(16) _Float16 shb[64 * LDB];
  const int o0 = blockIdx.y * 64;
  stage_weights(shb, wh + (size_t)o0 * DIMC);

  const int w  = threadIdx.x >> 5;
  const int t0 = blockIdx.x * 64 + w * 16;
  const _Float16* abase = ah + (size_t)t0 * DIMC;

  v8f acc[4] = {};
  v16h a = load_frag_a(abase, DIMC, 0);
#pragma unroll
  for (int kk = 0; kk < DIMC; kk += 32) {
    const v16h acur = a;
    if (kk + 32 < DIMC) a = load_frag_a(abase, DIMC, kk + 32);
#pragma unroll
    for (int t = 0; t < 4; ++t) {
      v16h bfr = load_frag_b(shb + 16 * t * LDB, LDB, kk);
      acc[t] = WMMA_F16(acur, bfr, acc[t]);
    }
  }

  const int l  = threadIdx.x & 31;
  const int n  = l & 15;
  const int mb = (l >> 4) * 8;
#pragma unroll
  for (int t = 0; t < 4; ++t) {
    const int o    = o0 + 16 * t + n;
    const float bv = bias[o];
#pragma unroll
    for (int r = 0; r < 8; ++r) {
      outp[(size_t)(t0 + mb + r) * DIMC + o] = acc[t][r] + bv;
    }
  }
}

// ---------------- launch ----------------
extern "C" void kernel_launch(void* const* d_in, const int* in_sizes, int n_in,
                              void* d_out, int out_size, void* d_ws, size_t ws_size,
                              hipStream_t stream) {
  const float* x      = (const float*)d_in[0];
  const float* qkv_w  = (const float*)d_in[1];
  const float* out_w  = (const float*)d_in[2];
  const float* out_b  = (const float*)d_in[3];
  const float* alphas = (const float*)d_in[4];
  const float* powers = (const float*)d_in[5];
  float* outp = (float*)d_out;

  char* ws = (char*)d_ws;
  const size_t MB = 1ull << 20;
  float*    maskb = (float*)(ws + 0);          //  4 MB  (1024x1024 f32)
  _Float16* xh    = (_Float16*)(ws + 4  * MB); //  8 MB  (8192x512)
  _Float16* wqkvh = (_Float16*)(ws + 12 * MB); // 1.5 MB (1536x512)
  _Float16* wouth = (_Float16*)(ws + 14 * MB); // 0.5 MB (512x512)
  _Float16* qh    = (_Float16*)(ws + 16 * MB); //  8 MB  (B,H,N,D)
  _Float16* kh    = (_Float16*)(ws + 24 * MB); //  8 MB  (B,H,N,D)
  _Float16* vth   = (_Float16*)(ws + 32 * MB); //  8 MB  (B,H,D,N)
  _Float16* attnh = (_Float16*)(ws + 40 * MB); //  8 MB  (8192x512)

  cvt_f32_f16<<<dim3(2048), dim3(256), 0, stream>>>(x, xh, TOKENS * DIMC);
  cvt_f32_f16<<<dim3(768),  dim3(256), 0, stream>>>(qkv_w, wqkvh, 3 * DIMC * DIMC);
  cvt_f32_f16<<<dim3(256),  dim3(256), 0, stream>>>(out_w, wouth, DIMC * DIMC);
  mask_kernel<<<dim3(1024), dim3(256), 0, stream>>>(alphas, powers, maskb);

  gemm_qkv<<<dim3(TOKENS / 64, (3 * DIMC) / 64), dim3(128), 0, stream>>>(
      xh, wqkvh, qh, kh, vth);

  attn_kernel<<<dim3(NP / 16, NB * NH), dim3(32), 0, stream>>>(
      qh, kh, vth, maskb, attnh);

  gemm_out<<<dim3(TOKENS / 64, DIMC / 64), dim3(128), 0, stream>>>(
      attnh, wouth, out_b, outp);
}
